// DistributionLossWithLabel_17008070493057
// MI455X (gfx1250) — matrix operand order repaired and checked by reading, not compiled, not run on hardware
//
#include <hip/hip_runtime.h>
#include <hip/hip_bf16.h>

typedef __attribute__((ext_vector_type(16))) _Float16 v16h;
typedef __attribute__((ext_vector_type(8)))  _Float16 v8h;
typedef __attribute__((ext_vector_type(8)))  float    v8f;

#define BN 8192
#define CN 64
#define NPAD 80                  // 64 p-cols + col 64 = a[j] + 15 zero cols
#define KSPLIT 4
#define KSEG (BN / KSPLIT)       // 2048

// float-element offsets into d_ws
#define OFF_LOGQ   0                           // BN*CN floats
#define OFF_POS    (BN * CN)                   // +BN
#define OFF_RLOSS  (OFF_POS + BN)              // +BN
#define OFF_PT     (OFF_RLOSS + BN)            // NPAD*BN halfs = NPAD*BN/2 floats
#define OFF_MACC   (OFF_PT + (NPAD * BN) / 2)  // +KSPLIT*BN*NPAD floats
#define WS_FLOATS  (OFF_MACC + KSPLIT * BN * NPAD)

// ---------------------------------------------------------------- stage -1: zero scratch
__global__ void kl_init_zero(float* __restrict__ ws) {
    const size_t n = (size_t)(WS_FLOATS - OFF_RLOSS);
    size_t idx = (size_t)blockIdx.x * blockDim.x + threadIdx.x;
    size_t stride = (size_t)gridDim.x * blockDim.x;
    for (size_t k = idx; k < n; k += stride) ws[OFF_RLOSS + k] = 0.0f;
}

// ---------------------------------------------------------------- stage 0: per-row prep
// logq (f32), positive[j], transposed fp16 panel pT[n][j]: rows 0..63 = p, row 64 = a[j]
__global__ void kl_stage0(const float* __restrict__ q, const float* __restrict__ p,
                          float* __restrict__ ws) {
    int j = blockIdx.x * blockDim.x + threadIdx.x;
    if (j >= BN) return;
    float* __restrict__ logq = ws + OFF_LOGQ;
    float* __restrict__ positive = ws + OFF_POS;
    _Float16* __restrict__ pT = (_Float16*)(ws + OFF_PT);

    const float* qr = q + (size_t)j * CN;
    const float* pr = p + (size_t)j * CN;
    float pos = 0.0f, a = 0.0f;
#pragma unroll 4
    for (int c = 0; c < CN; ++c) {
        float qv = qr[c], pv = pr[c];
        float lq = logf(qv), lp = logf(pv);
        logq[(size_t)j * CN + c] = lq;
        pos += pv * (lp - lq);
        a   += pv * lp;
        pT[(size_t)c * BN + j] = (_Float16)pv;      // coalesced across j
    }
    positive[j] = pos * (1.0f / CN);
    pT[(size_t)CN * BN + j] = (_Float16)(a * (1.0f / CN));   // row 64 = a[j]
}

// ---------------------------------------------------------------- stage 1: WMMA GEMM
// D = (2 - L) @ [p | a | 0]  accumulated in f32, per (row-tile, k-segment) wave.
__global__ void __launch_bounds__(256) kl_stage1(const float* __restrict__ L,
                                                 float* __restrict__ ws) {
    const _Float16* __restrict__ pT = (const _Float16*)(ws + OFF_PT);
    float* __restrict__ Macc = ws + OFF_MACC;

    const int lane = threadIdx.x & 31;
    const int wave = blockIdx.x * (blockDim.x >> 5) + (threadIdx.x >> 5); // 0..2047
    const int rt   = wave >> 2;              // row-tile 0..511
    const int kq   = wave & (KSPLIT - 1);    // k-segment 0..3
    const int R    = rt << 4;
    const int lg   = lane >> 4;              // lane group 0/1
    const int l15  = lane & 15;

    // A (16x32 f16): lane = row R+l15; halfs 0..15 = K akh..akh+15
    const float* __restrict__ Arow = L + (size_t)(R + l15) * BN + (size_t)kq * KSEG;
    const int akh = lg * 16;
    // B (32x16 f16): lane = col n (=l15 per tile); lanes 0-15 hold K 0..15, 16-31 hold K 16..31
    const int bkb = lg * 16;
    const _Float16* __restrict__ Bbase = pT + (size_t)kq * KSEG + bkb;

    v8f acc[5] = {};

    for (int ks = 0; ks < KSEG; ks += 32) {
        // ---- A fragment: convert (2 - L) to f16 on the fly ----
        const float* ap = Arow + ks + akh;
        __builtin_prefetch(ap + 512, 0, 1);          // -> global_prefetch_b8, stream L ahead
        v16h af;
#pragma unroll
        for (int e = 0; e < 8; ++e) af[e]     = (_Float16)(2.0f - ap[e]);
#pragma unroll
        for (int e = 0; e < 8; ++e) af[8 + e] = (_Float16)(2.0f - ap[8 + e]);

        // ---- 5 B fragments (N = 80) + WMMA ----
#pragma unroll
        for (int nt = 0; nt < 5; ++nt) {
            const v8h* bp = (const v8h*)(Bbase + (size_t)(nt * 16 + l15) * BN + ks);
            v8h b0 = bp[0], b1 = bp[1];              // 16 contiguous halfs (K bkb..bkb+15)
            v16h bf = __builtin_shufflevector(b0, b1, 0, 1, 2, 3, 4, 5, 6, 7,
                                                      8, 9, 10, 11, 12, 13, 14, 15);
            acc[nt] = __builtin_amdgcn_wmma_f32_16x16x32_f16(
                false, af, false, bf, (short)0, acc[nt], false, false);
        }
    }

    // ---- store partial tile to this k-segment's private slice (deterministic) ----
    float* __restrict__ outp = Macc + (size_t)kq * ((size_t)BN * NPAD);
#pragma unroll
    for (int nt = 0; nt < 5; ++nt) {
#pragma unroll
        for (int v = 0; v < 8; ++v) {
            int m = R + v + 8 * lg;                  // C/D layout: M = v + 8*laneGroup
            outp[(size_t)m * NPAD + nt * 16 + l15] = acc[nt][v];
        }
    }
}

// ---------------------------------------------------------------- stage 2: combine rows
__global__ void kl_stage2(float* __restrict__ ws) {
    int i = blockIdx.x * blockDim.x + threadIdx.x;
    if (i >= BN) return;
    const float* __restrict__ logq = ws + OFF_LOGQ;
    const float* __restrict__ positive = ws + OFF_POS;
    float* __restrict__ rowloss = ws + OFF_RLOSS;
    const float* __restrict__ Macc = ws + OFF_MACC;

    float dot = 0.0f;
    for (int c = 0; c < CN; ++c) {
        float m = 0.0f;
#pragma unroll
        for (int s = 0; s < KSPLIT; ++s)
            m += Macc[(size_t)s * BN * NPAD + (size_t)i * NPAD + c];
        dot += logq[(size_t)i * CN + c] * m;
    }
    float La = 0.0f;
#pragma unroll
    for (int s = 0; s < KSPLIT; ++s)
        La += Macc[(size_t)s * BN * NPAD + (size_t)i * NPAD + CN];

    float negative = La - dot * (1.0f / CN);
    rowloss[i] = positive[i] / negative;
}

// ---------------------------------------------------------------- stage 3: deterministic sum
__global__ void kl_stage3(const float* __restrict__ ws, float* __restrict__ out) {
    __shared__ float sm[256];
    const float* __restrict__ rowloss = ws + OFF_RLOSS;
    float s = 0.0f;
    for (int i = threadIdx.x; i < BN; i += 256) s += rowloss[i];
    sm[threadIdx.x] = s;
    __syncthreads();
    for (int off = 128; off > 0; off >>= 1) {
        if ((int)threadIdx.x < off) sm[threadIdx.x] += sm[threadIdx.x + off];
        __syncthreads();
    }
    if (threadIdx.x == 0) out[0] = sm[0];
}

// ---------------------------------------------------------------- launcher
extern "C" void kernel_launch(void* const* d_in, const int* in_sizes, int n_in,
                              void* d_out, int out_size, void* d_ws, size_t ws_size,
                              hipStream_t stream) {
    const float* q = (const float*)d_in[0];
    const float* p = (const float*)d_in[1];
    const float* L = (const float*)d_in[2];   // labels_matrix [B, B]
    float* out = (float*)d_out;
    float* ws  = (float*)d_ws;

    (void)in_sizes; (void)n_in; (void)out_size; (void)ws_size;

    kl_init_zero<<<512, 256, 0, stream>>>(ws);
    kl_stage0<<<BN / 256, 256, 0, stream>>>(q, p, ws);
    kl_stage1<<<(512 * KSPLIT) / 8, 256, 0, stream>>>(L, ws);  // 2048 waves, 8 per block
    kl_stage2<<<BN / 256, 256, 0, stream>>>(ws);
    kl_stage3<<<1, 256, 0, stream>>>(ws, out);
}
